// Mixer_63685775065491
// MI455X (gfx1250) — matrix operand order repaired and checked by reading, not compiled, not run on hardware
//
#include <hip/hip_runtime.h>
#include <hip/hip_bf16.h>

// ---------------- types & helpers ----------------
typedef __bf16 bf16_t;
typedef __attribute__((ext_vector_type(16))) __bf16 bf16x16;
typedef __attribute__((ext_vector_type(8)))  __bf16 bf16x8;
typedef __attribute__((ext_vector_type(2)))  __bf16 bf16x2;
typedef __attribute__((ext_vector_type(8)))  float  f32x8;

union bfcvt { unsigned short u; bf16_t b; };
union fucvt { float f; unsigned int u; };
union frag16 { bf16x16 v; bf16x8 h[2]; };
union b8x    { bf16x8 v; bf16x2 p[4]; };
union b2x    { unsigned u; bf16x2 v; };

// pack two f32 -> bf16x2.  Prefer the HW packed convert; else 3-VALU
// round+v_perm_b32 path (round-to-nearest, no tie fixup).
static __device__ inline bf16x2 tobf2(float lo, float hi) {
#if __has_builtin(__builtin_amdgcn_cvt_pk_bf16_f32)
  return __builtin_amdgcn_cvt_pk_bf16_f32(lo, hi);
#else
  fucvt f0; f0.f = lo; fucvt f1; f1.f = hi;
  unsigned r0 = f0.u + 0x7fffu;
  unsigned r1 = f1.u + 0x7fffu;
  // result bytes [0,1]=r0[2,3], [2,3]=r1[2,3]  (src0 bytes are 4..7)
  b2x r; r.u = __builtin_amdgcn_perm(r1, r0, 0x07060302u);
  return r.v;
#endif
}
static __device__ inline bf16_t tobf(float x) {
  fucvt f; f.f = x;
  bfcvt c; c.u = (unsigned short)((f.u + 0x7fffu) >> 16);
  return c.b;
}
static __device__ inline float frombf(bf16_t h) {
  bfcvt c; c.b = h;
  fucvt f; f.u = ((unsigned int)c.u) << 16;
  return f.f;
}

static __device__ inline f32x8 wmma_bf(bf16x16 a, bf16x16 b, f32x8 c) {
  // D = A(16x32 bf16) * B(32x16 bf16) + C(16x16 f32)
  return __builtin_amdgcn_wmma_f32_16x16x32_bf16(
      false, a, false, b, (short)0, c, false, false);
}

// A fragment: logical A[m][k] = base[m*ld + k]; k contiguous -> 2x ds_load_b128
static __device__ inline bf16x16 load_a_frag(const bf16_t* base, int ld,
                                             int row0, int k0, int lane) {
  const int m = row0 + (lane & 15);
  const int kh = (lane >> 4) * 8;
  frag16 f;
  f.h[0] = *(const bf16x8*)(base + m * ld + k0 + kh);
  f.h[1] = *(const bf16x8*)(base + m * ld + k0 + 16 + kh);
  return f.v;
}
// B fragment from k-contiguous storage: logical B[k][n] = base[n*ld + k]
static __device__ inline bf16x16 load_bT_frag(const bf16_t* base, int ld,
                                              int k0, int col0, int lane) {
  const int n = col0 + (lane & 15);
  const int kk = k0 + (lane >> 4) * 8;
  frag16 f;
  f.h[0] = *(const bf16x8*)(base + n * ld + kk);
  f.h[1] = *(const bf16x8*)(base + n * ld + kk + 16);
  return f.v;
}

// async 16B global -> LDS copy (gfx1250 async path, tracked by ASYNCcnt)
static __device__ inline void async_copy_b128(void* lds_dst, const void* gsrc) {
  unsigned lds = (unsigned)(unsigned long long)lds_dst;          // low 32b = LDS offset
  unsigned long long g = (unsigned long long)gsrc;
  asm volatile("global_load_async_to_lds_b128 %0, %1, off"
               :: "v"(lds), "v"(g) : "memory");
}
static __device__ inline void async_wait0() {
  asm volatile("s_wait_asynccnt 0x0" ::: "memory");
}

// ---------------- model dims ----------------
#define D_MODEL  2048
#define NHEADS   32
#define D_STATE  64
#define HEADDIM  64
#define D_INNER  2048
#define D_CONV   4
#define CHUNK    128
#define BATCH    2
#define SEQLEN   2048
#define NCHUNK   (SEQLEN / CHUNK)
#define CONV_DIM (D_INNER + 2 * NHEADS * D_STATE)      // 6144
#define PROJ_DIM (2 * D_INNER + 2 * NHEADS * D_STATE + NHEADS) // 8224
#define MROWS    (BATCH * SEQLEN)                      // 4096

// ---------------- kernel 1 & 8: tiled bf16-WMMA GEMM ----------------
// C[M,N] = A[M,K] (row-major f32) * Bm[N,K]^T (row-major f32)
// Requirement: M % TM == 0, K % TK == 0 (true for both call sites); only N tailed.
#define TM 128
#define TN 128
#define TK 32
#define LDA 40   // 80B rows: 16B-aligned 8-chunks
__global__ __launch_bounds__(256) void gemm_bf16_wmma(
    const float* __restrict__ A, const float* __restrict__ Bm,
    float* __restrict__ C, int M, int N, int K) {
  __shared__ __align__(16) bf16_t As[TM][LDA];   // [m][k]
  __shared__ __align__(16) bf16_t Bs[TN][LDA];   // [n][k]  (k contiguous)
  const int block_m = blockIdx.y * TM;
  const int block_n = blockIdx.x * TN;
  const int tid = threadIdx.x;
  const int wave = tid >> 5, lane = tid & 31;
  const int wm = (wave & 3) * 32;   // 4 M-slots of 32
  const int wn = (wave >> 2) * 64;  // 2 N-slots of 64
  f32x8 acc[2][4] = {};
  for (int k0 = 0; k0 < K; k0 += TK) {
    // A tile: 512 8-elem chunks; 2 per thread (no M guard: M % TM == 0)
    for (int c = tid; c < (TM * TK) / 8; c += 256) {
      int m = c >> 2, kc = (c & 3) * 8;
      const float4* gp = (const float4*)(A + (size_t)(block_m + m) * K + k0 + kc);
      float4 v0 = gp[0], v1 = gp[1];
      b8x o;
      o.p[0] = tobf2(v0.x, v0.y); o.p[1] = tobf2(v0.z, v0.w);
      o.p[2] = tobf2(v1.x, v1.y); o.p[3] = tobf2(v1.z, v1.w);
      *(bf16x8*)&As[m][kc] = o.v;
    }
    // B tile: [n][k], global Bm[n][k] is k-contiguous -> fully coalesced
    for (int c = tid; c < (TN * TK) / 8; c += 256) {
      int n = c >> 2, kc = (c & 3) * 8;
      int gn = block_n + n;
      float4 v0 = make_float4(0.f, 0.f, 0.f, 0.f), v1 = v0;
      if (gn < N) {
        const float4* gp = (const float4*)(Bm + (size_t)gn * K + k0 + kc);
        v0 = gp[0]; v1 = gp[1];
      }
      b8x o;
      o.p[0] = tobf2(v0.x, v0.y); o.p[1] = tobf2(v0.z, v0.w);
      o.p[2] = tobf2(v1.x, v1.y); o.p[3] = tobf2(v1.z, v1.w);
      *(bf16x8*)&Bs[n][kc] = o.v;
    }
    __syncthreads();
    bf16x16 afrag[2];
#pragma unroll
    for (int i = 0; i < 2; ++i)
      afrag[i] = load_a_frag(&As[0][0], LDA, wm + i * 16, 0, lane);
#pragma unroll
    for (int j = 0; j < 4; ++j) {
      bf16x16 bfr = load_bT_frag(&Bs[0][0], LDA, 0, wn + j * 16, lane);
#pragma unroll
      for (int i = 0; i < 2; ++i)
        acc[i][j] = wmma_bf(afrag[i], bfr, acc[i][j]);
    }
    __syncthreads();
  }
  const int nlane = lane & 15, mbase = (lane >> 4) * 8;
#pragma unroll
  for (int i = 0; i < 2; ++i)
#pragma unroll
    for (int j = 0; j < 4; ++j) {
      int gn = block_n + wn + j * 16 + nlane;
      if (gn >= N) continue;
#pragma unroll
      for (int v = 0; v < 8; ++v) {
        int gm = block_m + wm + i * 16 + mbase + v;
        C[(size_t)gm * N + gn] = acc[i][j][v];
      }
    }
}

// ---------------- kernel 2: depthwise conv + SiLU, scatter to chunk tiles ----
// X is stored transposed per chunk: [b,h,c][p][t]  (t contiguous, k-dim of WMMA)
__global__ void conv_act_kernel(const float* __restrict__ proj,
                                const float* __restrict__ conv_w,
                                const float* __restrict__ conv_b,
                                bf16_t* __restrict__ Xbf,
                                bf16_t* __restrict__ Bbf,
                                bf16_t* __restrict__ Cbf) {
  size_t gid = (size_t)blockIdx.x * blockDim.x + threadIdx.x;
  if (gid >= (size_t)BATCH * SEQLEN * CONV_DIM) return;
  int cd = (int)(gid % CONV_DIM);
  int l  = (int)((gid / CONV_DIM) % SEQLEN);
  int b  = (int)(gid / ((size_t)CONV_DIM * SEQLEN));
  float v = conv_b[cd];
#pragma unroll
  for (int w = 0; w < D_CONV; ++w) {
    int lp = l + w - (D_CONV - 1);
    if (lp >= 0)
      v += proj[((size_t)(b * SEQLEN + lp)) * PROJ_DIM + cd] * conv_w[cd * D_CONV + w];
  }
  float s = v / (1.f + __expf(-v));     // SiLU
  bf16_t sv = tobf(s);
  int cix = l >> 7, t = l & 127;
  if (cd < D_INNER) {
    int h = cd >> 6, p = cd & 63;
    Xbf[(((size_t)(b * NHEADS + h) * NCHUNK + cix) * HEADDIM + p) * CHUNK + t] = sv;
  } else if (cd < D_INNER + NHEADS * D_STATE) {
    int q = cd - D_INNER, h = q >> 6, n = q & 63;
    Bbf[(((size_t)(b * NHEADS + h) * NCHUNK + cix) * CHUNK + t) * D_STATE + n] = sv;
  } else {
    int q = cd - D_INNER - NHEADS * D_STATE, h = q >> 6, n = q & 63;
    Cbf[(((size_t)(b * NHEADS + h) * NCHUNK + cix) * CHUNK + t) * D_STATE + n] = sv;
  }
}

// ---------------- kernel 3: A = -softplus(proj tail) -> chunk layout --------
__global__ void alog_kernel(const float* __restrict__ proj,
                            float* __restrict__ dtbuf) {
  int gid = blockIdx.x * blockDim.x + threadIdx.x;
  if (gid >= BATCH * SEQLEN * NHEADS) return;
  int h = gid & 31;
  int l = (gid >> 5) & (SEQLEN - 1);
  int b = gid >> 16;                       // 2048*32 = 65536
  float x = proj[((size_t)(b * SEQLEN + l)) * PROJ_DIM + (PROJ_DIM - NHEADS) + h];
  float sp = (x > 20.f) ? x : log1pf(__expf(x));
  dtbuf[(((size_t)(b * NHEADS + h) * NCHUNK + (l >> 7)) * CHUNK) + (l & 127)] = -sp;
}

// ---------------- kernel 4: per-chunk SSD (all-WMMA, async LDS fill) ---------
__global__ __launch_bounds__(256) void chunk_kernel(
    const bf16_t* __restrict__ Xbf, const bf16_t* __restrict__ Bbf,
    const bf16_t* __restrict__ Cbf, const float* __restrict__ dtbuf,
    float* __restrict__ cumAbuf, float* __restrict__ Ydiag,
    float* __restrict__ statesL) {
  const int blk = blockIdx.x;            // (b*NHEADS + h)*NCHUNK + c
  const size_t tileoff = (size_t)blk * CHUNK * 64;
  __shared__ __align__(16) bf16_t Cs[CHUNK][72];        // [t][n] (A of G)
  __shared__ __align__(16) bf16_t Bs2[CHUNK][72];       // [t][n] (B of G, k=n contig)
  __shared__ __align__(16) bf16_t Bs2T[64][CHUNK + 8];  // [n][t] (A of states, k=t contig)
  __shared__ __align__(16) bf16_t XsT[64][CHUNK + 8];   // [p][t] (B operand, k=t contig)
  __shared__ __align__(16) bf16_t Gs[CHUNK][CHUNK + 8]; // [t][s] (A of Y_diag)
  __shared__ float  cumA[CHUNK];
  const int tid = threadIdx.x, lane = tid & 31, wave = tid >> 5;

  // ---- async DMA of the three 16KB bf16 tiles straight into LDS
  {
    const bf16_t* Cg = Cbf + tileoff;
    const bf16_t* Bg = Bbf + tileoff;
    const bf16_t* Xg = Xbf + tileoff;   // global X is [p][t]
    for (int c = tid; c < 1024; c += 256) {             // 16B chunks
      int t = c >> 3, n0 = (c & 7) * 8;                 // C/B: [t][n]
      int p = c >> 4, t0 = (c & 15) * 8;                // X:   [p][t]
      async_copy_b128(&Cs[t][n0],  Cg + c * 8);
      async_copy_b128(&Bs2[t][n0], Bg + c * 8);
      async_copy_b128(&XsT[p][t0], Xg + c * 8);
    }
    async_wait0();
  }
  if (tid < CHUNK) cumA[tid] = dtbuf[(size_t)blk * CHUNK + tid];
  __syncthreads();
  // ---- build transposed B copy (LDS -> LDS) for the state GEMM A-operand
  for (int c = tid; c < 1024; c += 256) {
    int t = c >> 3, n0 = (c & 7) * 8;
    bf16x8 vB = *(const bf16x8*)&Bs2[t][n0];
#pragma unroll
    for (int e = 0; e < 8; ++e) Bs2T[n0 + e][t] = vB[e];
  }
  // inclusive scan (Hillis–Steele) over 128 A values
  for (int off = 1; off < CHUNK; off <<= 1) {
    float add = (tid < CHUNK && tid >= off) ? cumA[tid - off] : 0.f;
    __syncthreads();
    if (tid < CHUNK) cumA[tid] += add;
    __syncthreads();
  }
  if (tid < CHUNK) cumAbuf[(size_t)blk * CHUNK + tid] = cumA[tid];
  __syncthreads();

  // ---- G = C(128x64) * B^T(64x128), masked by L, -> Gs (bf16)
  {
    const int row0 = wave * 16;
    f32x8 acc[8] = {};
    for (int kt = 0; kt < 64; kt += 32) {
      bf16x16 af = load_a_frag(&Cs[0][0], 72, row0, kt, lane);
#pragma unroll
      for (int j = 0; j < 8; ++j) {
        bf16x16 bfr = load_bT_frag(&Bs2[0][0], 72, kt, j * 16, lane);
        acc[j] = wmma_bf(af, bfr, acc[j]);
      }
    }
    const int sl = lane & 15, mb = (lane >> 4) * 8;
#pragma unroll
    for (int j = 0; j < 8; ++j) {
      int s = j * 16 + sl;
#pragma unroll
      for (int v = 0; v < 8; ++v) {
        int t = row0 + mb + v;
        float lm = (s <= t) ? __expf(cumA[t] - cumA[s]) : 0.f;
        Gs[t][s] = tobf(acc[j][v] * lm);
      }
    }
  }
  __syncthreads();

  // ---- Y_diag = (G*L)(128x128) * X(128x64)
  {
    const int row0 = wave * 16;
    f32x8 acc[4] = {};
    for (int kt = 0; kt < CHUNK; kt += 32) {
      bf16x16 af = load_a_frag(&Gs[0][0], CHUNK + 8, row0, kt, lane);
#pragma unroll
      for (int j = 0; j < 4; ++j) {
        bf16x16 bfr = load_bT_frag(&XsT[0][0], CHUNK + 8, kt, j * 16, lane);
        acc[j] = wmma_bf(af, bfr, acc[j]);
      }
    }
    const int pl = lane & 15, mb = (lane >> 4) * 8;
#pragma unroll
    for (int j = 0; j < 4; ++j)
#pragma unroll
      for (int v = 0; v < 8; ++v) {
        int t = row0 + mb + v, p = j * 16 + pl;
        Ydiag[tileoff + t * 64 + p] = acc[j][v];
      }
  }
  __syncthreads();

  // ---- decay X in place: Xd[p][t] = X[p][t] * exp(cumA_last - cumA[t])
  {
    float alast = cumA[CHUNK - 1];
    for (int i = tid; i < (CHUNK * 64) / 2; i += 256) {
      int p = i >> 6, t0 = (i & 63) * 2;
      bf16x2 v = *(bf16x2*)&XsT[p][t0];
      float f0 = frombf(v[0]) * __expf(alast - cumA[t0]);
      float f1 = frombf(v[1]) * __expf(alast - cumA[t0 + 1]);
      *(bf16x2*)&XsT[p][t0] = tobf2(f0, f1);
    }
  }
  __syncthreads();

  // ---- states = B^T(64x128) * Xd(128x64) -> 64x64
  {
    const int mrow = (wave & 3) * 16;
    const int n0 = (wave >> 2) * 32;
    f32x8 acc[2] = {};
    for (int kt = 0; kt < CHUNK; kt += 32) {
      bf16x16 af = load_a_frag(&Bs2T[0][0], CHUNK + 8, mrow, kt, lane);
#pragma unroll
      for (int j = 0; j < 2; ++j) {
        bf16x16 bfr = load_bT_frag(&XsT[0][0], CHUNK + 8, kt, n0 + j * 16, lane);
        acc[j] = wmma_bf(af, bfr, acc[j]);
      }
    }
    const int pl = lane & 15, mb = (lane >> 4) * 8;
#pragma unroll
    for (int j = 0; j < 2; ++j)
#pragma unroll
      for (int v = 0; v < 8; ++v) {
        int n = mrow + mb + v, p = n0 + j * 16 + pl;
        statesL[(size_t)blk * 4096 + n * 64 + p] = acc[j][v];
      }
  }
}

// ---------------- kernel 5: inter-chunk state recurrence ---------------------
__global__ __launch_bounds__(256) void chunkscan_kernel(
    const float* __restrict__ statesL, const float* __restrict__ cumAbuf,
    float* __restrict__ statesIn) {
  const int bh = blockIdx.x;             // b*NHEADS + h
  const int tid = threadIdx.x;
  float carry[16];
#pragma unroll
  for (int i = 0; i < 16; ++i) carry[i] = 0.f;
  for (int c = 0; c < NCHUNK; ++c) {
    size_t base = ((size_t)bh * NCHUNK + c) * 4096;
#pragma unroll
    for (int i = 0; i < 16; ++i) statesIn[base + i * 256 + tid] = carry[i];
    float dec = __expf(cumAbuf[((size_t)bh * NCHUNK + c) * CHUNK + (CHUNK - 1)]);
#pragma unroll
    for (int i = 0; i < 16; ++i)
      carry[i] = carry[i] * dec + statesL[base + i * 256 + tid];
  }
}

// ---------------- kernel 6: Y_off (WMMA) + combine ---------------------------
__global__ __launch_bounds__(256) void yoff_kernel(
    const bf16_t* __restrict__ Cbf, const bf16_t* __restrict__ Xbf,
    const float* __restrict__ cumAbuf, const float* __restrict__ statesIn,
    const float* __restrict__ Ydiag, const float* __restrict__ Dvec,
    float* __restrict__ y) {
  const int blk = blockIdx.x;            // (b*NHEADS + h)*NCHUNK + c
  const int h = (blk >> 4) & 31;
  const int b = blk >> 9;
  const int c = blk & 15;
  const size_t tileoff = (size_t)blk * CHUNK * 64;
  __shared__ __align__(16) bf16_t Ct[CHUNK][72];  // [t][n], A operand
  __shared__ __align__(16) bf16_t SbT[64][72];    // [p][n], B operand (k=n contig)
  __shared__ float  ca[CHUNK];
  const int tid = threadIdx.x, lane = tid & 31, wave = tid >> 5;
  if (tid < CHUNK) ca[tid] = cumAbuf[(size_t)blk * CHUNK + tid];
  __syncthreads();
  {
    const bf16x8* Cg = (const bf16x8*)(Cbf + tileoff);
    for (int cc = tid; cc < 1024; cc += 256) {
      int t = cc >> 3, n0 = (cc & 7) * 8;
      bf16x8 v = Cg[cc];
      float sc = __expf(ca[t]);
      b8x o;
#pragma unroll
      for (int e = 0; e < 4; ++e)
        o.p[e] = tobf2(frombf(v[2 * e]) * sc, frombf(v[2 * e + 1]) * sc);
      *(bf16x8*)&Ct[t][n0] = o.v;
    }
    for (int i = tid; i < 64 * 64; i += 256) {
      int n = i >> 6, p = i & 63;
      SbT[p][n] = tobf(statesIn[(size_t)blk * 4096 + i]);
    }
  }
  __syncthreads();
  const int row0 = wave * 16;
  f32x8 acc[4] = {};
  for (int kt = 0; kt < 64; kt += 32) {
    bf16x16 af = load_a_frag(&Ct[0][0], 72, row0, kt, lane);
#pragma unroll
    for (int j = 0; j < 4; ++j) {
      bf16x16 bfr = load_bT_frag(&SbT[0][0], 72, kt, j * 16, lane);
      acc[j] = wmma_bf(af, bfr, acc[j]);
    }
  }
  const float Dh = Dvec[h];
  const int pl = lane & 15, mb = (lane >> 4) * 8;
#pragma unroll
  for (int j = 0; j < 4; ++j)
#pragma unroll
    for (int v = 0; v < 8; ++v) {
      int t = row0 + mb + v, p = j * 16 + pl;
      float val = acc[j][v] + Ydiag[tileoff + t * 64 + p] +
                  Dh * frombf(Xbf[tileoff + p * CHUNK + t]);   // X is [p][t]
      int l = c * CHUNK + t;
      y[((size_t)(b * SEQLEN + l)) * D_INNER + h * HEADDIM + p] = val;
    }
}

// ---------------- kernel 7: SiLU gate + RMSNorm ------------------------------
__global__ __launch_bounds__(256) void gate_rms_kernel(
    const float* __restrict__ y, const float* __restrict__ proj,
    const float* __restrict__ z_bias, const float* __restrict__ rms_w,
    float* __restrict__ yn) {
  const int row = blockIdx.x;            // b*SEQLEN + l
  const float* yrow = y + (size_t)row * D_INNER;
  const float* zrow = proj + (size_t)row * PROJ_DIM + CONV_DIM;
  __shared__ float red[256];
  float vals[8];
  float ss = 0.f;
#pragma unroll
  for (int i = 0; i < 8; ++i) {
    int e = i * 256 + threadIdx.x;
    float z = zrow[e] + z_bias[e];
    float g = z / (1.f + __expf(-z));
    float v = yrow[e] * g;
    vals[i] = v;
    ss += v * v;
  }
  red[threadIdx.x] = ss;
  __syncthreads();
  for (int off = 128; off > 0; off >>= 1) {
    if (threadIdx.x < off) red[threadIdx.x] += red[threadIdx.x + off];
    __syncthreads();
  }
  float inv = rsqrtf(red[0] / (float)D_INNER + 1e-5f);
#pragma unroll
  for (int i = 0; i < 8; ++i) {
    int e = i * 256 + threadIdx.x;
    yn[(size_t)row * D_INNER + e] = vals[i] * inv * rms_w[e];
  }
}

// ---------------- host-side launch ------------------------------------------
extern "C" void kernel_launch(void* const* d_in, const int* in_sizes, int n_in,
                              void* d_out, int out_size, void* d_ws, size_t ws_size,
                              hipStream_t stream) {
  (void)in_sizes; (void)n_in; (void)out_size; (void)ws_size;
  const float* u      = (const float*)d_in[0];
  const float* W_in   = (const float*)d_in[1];
  const float* z_bias = (const float*)d_in[2];
  const float* conv_w = (const float*)d_in[3];
  const float* conv_b = (const float*)d_in[4];
  const float* Dvec   = (const float*)d_in[5];
  const float* rms_w  = (const float*)d_in[6];
  const float* W_out  = (const float*)d_in[7];
  float* out = (float*)d_out;

  // workspace carve-up (all sizes are multiples of 256B)
  char* p = (char*)d_ws;
  float* proj = (float*)p;      p += (size_t)MROWS * PROJ_DIM * 4;          // 134.7 MB
  bf16_t* Xbf = (bf16_t*)p;     p += (size_t)BATCH * SEQLEN * D_INNER * 2;  // 16.8 MB
  bf16_t* Bbf = (bf16_t*)p;     p += (size_t)BATCH * SEQLEN * NHEADS * D_STATE * 2;
  bf16_t* Cbf = (bf16_t*)p;     p += (size_t)BATCH * SEQLEN * NHEADS * D_STATE * 2;
  float* dtbuf = (float*)p;     p += (size_t)BATCH * NHEADS * SEQLEN * 4;   // 0.5 MB
  float* cumAbuf = (float*)p;   p += (size_t)BATCH * NHEADS * SEQLEN * 4;
  float* Ydiag = (float*)p;     p += (size_t)BATCH * SEQLEN * D_INNER * 4;  // 33.6 MB
  float* statesL = (float*)p;   p += (size_t)BATCH * NHEADS * NCHUNK * 4096 * 4;
  float* statesIn = (float*)p;  p += (size_t)BATCH * NHEADS * NCHUNK * 4096 * 4;
  float* ybuf = (float*)p;      p += (size_t)MROWS * D_INNER * 4;           // 33.6 MB
  float* ynbuf = (float*)p;     p += (size_t)MROWS * D_INNER * 4;

  // 1) in-projection GEMM: proj[4096,8224] = u[4096,2048] * W_in^T
  {
    dim3 grid((PROJ_DIM + TN - 1) / TN, MROWS / TM);
    gemm_bf16_wmma<<<grid, 256, 0, stream>>>(u, W_in, proj, MROWS, PROJ_DIM, D_MODEL);
  }
  // 2) conv + SiLU, scatter to chunk-major bf16 tiles
  {
    size_t total = (size_t)BATCH * SEQLEN * CONV_DIM;
    conv_act_kernel<<<(unsigned)((total + 255) / 256), 256, 0, stream>>>(
        proj, conv_w, conv_b, Xbf, Bbf, Cbf);
  }
  // 3) A = -softplus
  alog_kernel<<<(BATCH * SEQLEN * NHEADS) / 256, 256, 0, stream>>>(proj, dtbuf);
  // 4) per-chunk SSD
  chunk_kernel<<<BATCH * NHEADS * NCHUNK, 256, 0, stream>>>(
      Xbf, Bbf, Cbf, dtbuf, cumAbuf, Ydiag, statesL);
  // 5) inter-chunk recurrence
  chunkscan_kernel<<<BATCH * NHEADS, 256, 0, stream>>>(statesL, cumAbuf, statesIn);
  // 6) Y_off + combine -> y[b,l,e]
  yoff_kernel<<<BATCH * NHEADS * NCHUNK, 256, 0, stream>>>(
      Cbf, Xbf, cumAbuf, statesIn, Ydiag, Dvec, ybuf);
  // 7) gate + RMSNorm
  gate_rms_kernel<<<MROWS, 256, 0, stream>>>(ybuf, proj, z_bias, rms_w, ynbuf);
  // 8) out-projection GEMM: out[4096,2048] = yn[4096,2048] * W_out^T
  {
    dim3 grid(D_MODEL / TN, MROWS / TM);
    gemm_bf16_wmma<<<grid, 256, 0, stream>>>(ynbuf, W_out, out, MROWS, D_MODEL, D_INNER);
  }
}
